// rnnBase_43061342109708
// MI455X (gfx1250) — matrix operand order repaired and checked by reading, not compiled, not run on hardware
//
#include <hip/hip_runtime.h>
#include <hip/hip_bf16.h>

typedef __attribute__((ext_vector_type(16))) _Float16 v16h;
typedef __attribute__((ext_vector_type(8)))  _Float16 v8h;
typedef __attribute__((ext_vector_type(8)))  float    v8f;
typedef __attribute__((ext_vector_type(4)))  unsigned int u32x4;
typedef __attribute__((ext_vector_type(8)))  int      i32x8;
typedef __attribute__((ext_vector_type(4)))  int      i32x4;

#define HID   1024
#define H3    3072
#define SEQ   512
#define VOCAB 32000
#define GRU_NWG 64

// LDS A-tile row pitch: 32 f16 data + 8 f16 TDM pad = 40 f16 (80B) per row.
// 20 DWORDs/row -> lanes 0..15 hit 16 distinct banks on ds_load_b128.
#define A_PITCH 40

// ---------------------------------------------------------------------------
// f32 -> f16 bulk convert (vectorized by float4)
// ---------------------------------------------------------------------------
__global__ void cvt_f32_f16(const float* __restrict__ src,
                            _Float16* __restrict__ dst, long n4) {
    long i = (long)blockIdx.x * blockDim.x + threadIdx.x;
    if (i >= n4) return;
    float4 v = ((const float4*)src)[i];
    _Float16* o = dst + i * 4;
    o[0] = (_Float16)v.x; o[1] = (_Float16)v.y;
    o[2] = (_Float16)v.z; o[3] = (_Float16)v.w;
}

// ---------------------------------------------------------------------------
// Embedding gather + f16 convert. isDec: token(t) = (t==0) ? 0 : tok[t-1]
// ---------------------------------------------------------------------------
__global__ void gather_embed(const int* __restrict__ tok,
                             const float* __restrict__ emb,
                             _Float16* __restrict__ out, int isDec) {
    int t = blockIdx.x;
    int token = isDec ? (t == 0 ? 0 : tok[t - 1]) : tok[t];
    const float4* src = (const float4*)(emb + (size_t)token * HID);
    float4 v = src[threadIdx.x];
    _Float16* o = out + (size_t)t * HID + (size_t)threadIdx.x * 4;
    o[0] = (_Float16)v.x; o[1] = (_Float16)v.y;
    o[2] = (_Float16)v.z; o[3] = (_Float16)v.w;
}

// ---------------------------------------------------------------------------
// WMMA fragment loads (ISA 7.12.2, 16-bit operands, wave32)
//   A 16x32: lane = (m, hi): elems 0..7 -> K hi*8..hi*8+7; 8..15 -> +16
//   B 32x16: lane = (n, hi): elems 0..7 -> K hi*8..hi*8+7; 8..15 -> +16
// ---------------------------------------------------------------------------
__device__ inline v16h fragA_from(const _Float16* p) {         // p = row + hi*8
    v8h lo = *(const v8h*)p;
    v8h hh = *(const v8h*)(p + 16);
    return __builtin_shufflevector(lo, hh, 0,1,2,3,4,5,6,7,8,9,10,11,12,13,14,15);
}
__device__ inline v16h fragB_from(const _Float16* p) {         // p = row + hi*8
    v8h lo = *(const v8h*)p;
    v8h hh = *(const v8h*)(p + 16);
    return __builtin_shufflevector(lo, hh, 0,1,2,3,4,5,6,7,8,9,10,11,12,13,14,15);
}

// ---------------------------------------------------------------------------
// TDM: async-load a 64-row x 32-col f16 A-tile (row-major, row stride K) into
// LDS at byte offset ldsOff, padding each 64B row to 80B. D# per ISA ch.8:
//   group0: count=1, lds_addr, global_addr(57b), type=2
//   group1: data_size=1(2B), pad_en, pad_interval=16DW, pad_amount=4DW,
//           tensor_dim0=K, tensor_dim1=M, tile_dim0=32, tile_dim1=64,
//           tensor_dim0_stride=K
// Tracked by TENSORcnt (issued by wave 0 only; EXEC ignored by TDM).
// ---------------------------------------------------------------------------
__device__ inline void tdm_load_tileA(const _Float16* A, int M, int K,
                                      int rowBase, int kk, unsigned ldsOff) {
    unsigned long long ga =
        (unsigned long long)(size_t)(A + (size_t)rowBase * K + kk);
    u32x4 g0;
    g0.x = 1u;                                        // count=1, user descriptor
    g0.y = ldsOff;                                    // lds_addr (bytes)
    g0.z = (unsigned)ga;                              // global_addr[31:0]
    g0.w = (unsigned)((ga >> 32) & 0x1FFFFFFull)      // global_addr[56:32]
         | (2u << 30);                                // type=2 (image)
    i32x8 g1;
    g1[0] = (1 << 16)                                 // data_size = 2 bytes
          | (1 << 20)                                 // pad_enable
          | (3 << 22)                                 // pad_interval: 16 DWORDs
          | (3 << 25);                                // pad_amount: 4 DWORDs
    g1[1] = (int)((unsigned)K << 16);                 // tensor_dim0 lo16
    g1[2] = (int)(((unsigned)K >> 16) | ((unsigned)M << 16));  // dim0 hi | dim1 lo
    g1[3] = (int)(((unsigned)M >> 16) | (32u << 16)); // dim1 hi | tile_dim0=32
    g1[4] = 64;                                       // tile_dim1 = 64 rows
    g1[5] = K;                                        // tensor_dim0_stride lo32
    g1[6] = 0;
    g1[7] = 0;
    i32x4 z4 = {0, 0, 0, 0};
    i32x8 z8 = {0, 0, 0, 0, 0, 0, 0, 0};
    __builtin_amdgcn_tensor_load_to_lds(g0, g1, z4, z4, z8, 0);
}

// ---------------------------------------------------------------------------
// C[M,N] = A[M,K] * B[N,K]^T + bias[N]    (A,B f16, C f32)
// Block tile 64(M) x 256(N); 8 waves (2x4); wave tile 32x64 = 2x4 WMMA frags.
// A tile staged to LDS by the Tensor Data Mover (double-buffered, 1-deep
// issue-ahead); B fragments streamed from global/L2. K multiple of 32.
// ---------------------------------------------------------------------------
__global__ void __launch_bounds__(256)
gemm_wmma_f16(const _Float16* __restrict__ A, const _Float16* __restrict__ B,
              const float* __restrict__ bias, float* __restrict__ C,
              int M, int N, int K) {
    __shared__ __align__(16) _Float16 shA[2][64 * A_PITCH];
    const int tid  = threadIdx.x;
    const int wave = tid >> 5;
    const int lane = tid & 31;
    const int wm = wave >> 2;                    // 0..1
    const int wn = wave & 3;                     // 0..3
    const int mblk = blockIdx.y * 64;
    const int m0 = mblk + wm * 32;
    const int n0 = blockIdx.x * 256 + wn * 64;
    const int fm = lane & 15;
    const int hi = lane >> 4;

    const unsigned lds0 = (unsigned)(size_t)&shA[0][0];
    const unsigned ldsTileBytes = 64 * A_PITCH * 2;

    // B pointers: 4 frags of 16 columns each
    const _Float16* bp[4];
#pragma unroll
    for (int j = 0; j < 4; ++j)
        bp[j] = B + (size_t)(n0 + 16 * j + fm) * K + hi * 8;

    v8f acc[2][4] = {};

    if (wave == 0) tdm_load_tileA(A, M, K, mblk, 0, lds0);

    const int NIT = K / 32;
    for (int it = 0; it < NIT; ++it) {
        const int kk  = it * 32;
        const int cur = it & 1;
        __syncthreads();   // everyone done reading shA[cur^1] (prev iter)
        if (wave == 0) {
            if (it + 1 < NIT) {
                tdm_load_tileA(A, M, K, mblk, kk + 32,
                               lds0 + (unsigned)(cur ^ 1) * ldsTileBytes);
                __builtin_amdgcn_s_wait_tensorcnt(1);   // current tile done
            } else {
                __builtin_amdgcn_s_wait_tensorcnt(0);   // drain
            }
        }
        __syncthreads();   // shA[cur] now valid for all waves

        __builtin_prefetch(bp[0] + kk + 128, 0, 1);   // global_prefetch_b8
        __builtin_prefetch(bp[2] + kk + 128, 0, 1);

        v16h a[2], b[4];
#pragma unroll
        for (int i = 0; i < 2; ++i)
            a[i] = fragA_from(&shA[cur][(wm * 32 + 16 * i + fm) * A_PITCH + hi * 8]);
#pragma unroll
        for (int j = 0; j < 4; ++j)
            b[j] = fragB_from(bp[j] + kk);

#pragma unroll
        for (int i = 0; i < 2; ++i)
#pragma unroll
            for (int j = 0; j < 4; ++j)
                acc[i][j] = __builtin_amdgcn_wmma_f32_16x16x32_f16(
                    false, a[i], false, b[j], (short)0, acc[i][j], false, false);
    }

    // C frag: lanes 0-15 -> M = r, lanes 16-31 -> M = 8+r; N = lane&15
#pragma unroll
    for (int i = 0; i < 2; ++i)
#pragma unroll
        for (int j = 0; j < 4; ++j) {
            const int col = n0 + 16 * j + fm;
            const float bv = bias[col];
#pragma unroll
            for (int r = 0; r < 8; ++r) {
                const int row = m0 + 16 * i + r + 8 * hi;
                C[(size_t)row * N + col] = acc[i][j][r] + bv;
            }
        }
}

// ---------------------------------------------------------------------------
// Grid barrier for the persistent GRU kernel (counter memset per launch).
// ---------------------------------------------------------------------------
__device__ inline void grid_sync(unsigned int* bar, unsigned int target) {
    __syncthreads();
    if (threadIdx.x == 0) {
        __threadfence();
        atomicAdd(bar, 1u);
        while (__atomic_load_n(bar, __ATOMIC_RELAXED) < target)
            __builtin_amdgcn_s_sleep(1);
        __threadfence();
    }
    __syncthreads();
}

// ---------------------------------------------------------------------------
// Persistent fused GRU: 512 encoder steps then 512 decoder steps.
// gx = Wih@x + bih precomputed. Each of 64 WGs owns 16 hidden indices ->
// 48 Whh rows per step, fused gate epilogue; only the 4KB h vector crosses
// WGs (double-buffered). Decoder h also stored as f16 for the WMMA proj.
// ---------------------------------------------------------------------------
__global__ void __launch_bounds__(256)
gru_persistent(const float* __restrict__ GXenc, const float* __restrict__ GXdec,
               const float* __restrict__ WhhEnc, const float* __restrict__ WhhDec,
               const float* __restrict__ bhhEnc, const float* __restrict__ bhhDec,
               float* __restrict__ hbuf, _Float16* __restrict__ decHsH,
               unsigned int* __restrict__ bar) {
    __shared__ float sh_h[HID];
    __shared__ float sh_gh[48];
    const int tid  = threadIdx.x;
    const int wave = tid >> 5;
    const int lane = tid & 31;
    const int base = blockIdx.x * (HID / GRU_NWG);

    if (tid < 16) hbuf[base + tid] = 0.0f;            // h0 = 0
    unsigned int target = GRU_NWG;
    grid_sync(bar, target);

    for (int step = 0; step < 2 * SEQ; ++step) {
        const float* hin  = hbuf + (step & 1) * HID;
        float*       hout = hbuf + ((step + 1) & 1) * HID;
        const bool enc = step < SEQ;
        const float* gx  = enc ? GXenc + (size_t)step * H3
                               : GXdec + (size_t)(step - SEQ) * H3;
        const float* Whh = enc ? WhhEnc : WhhDec;
        const float* bhh = enc ? bhhEnc : bhhDec;

        for (int i = tid; i < HID; i += 256) sh_h[i] = hin[i];
        __syncthreads();

        for (int rr = wave; rr < 48; rr += 8) {
            const int g   = rr >> 4;
            const int il  = rr & 15;
            const int row = g * HID + base + il;
            const float* w = Whh + (size_t)row * HID;
            float acc = 0.0f;
#pragma unroll 4
            for (int j = lane; j < HID; j += 32) acc += w[j] * sh_h[j];
#pragma unroll
            for (int off = 16; off > 0; off >>= 1)
                acc += __shfl_xor(acc, off, 32);
            if (lane == 0) sh_gh[rr] = acc + bhh[row];
        }
        __syncthreads();

        if (tid < 16) {
            const int gi = base + tid;
            const float r = 1.0f / (1.0f + __expf(-(gx[gi]       + sh_gh[tid])));
            const float z = 1.0f / (1.0f + __expf(-(gx[HID + gi] + sh_gh[16 + tid])));
            const float n = tanhf(gx[2 * HID + gi] + r * sh_gh[32 + tid]);
            const float hv = (1.0f - z) * n + z * sh_h[gi];
            hout[gi] = hv;
            if (!enc) decHsH[(size_t)(step - SEQ) * HID + gi] = (_Float16)hv;
        }
        target += GRU_NWG;
        grid_sync(bar, target);
    }
}

// ---------------------------------------------------------------------------
extern "C" void kernel_launch(void* const* d_in, const int* in_sizes, int n_in,
                              void* d_out, int out_size, void* d_ws, size_t ws_size,
                              hipStream_t stream) {
    const int*   inputs  = (const int*)d_in[0];
    const int*   targets = (const int*)d_in[1];
    const float* emb     = (const float*)d_in[2];
    const float* encWih  = (const float*)d_in[3];
    const float* encWhh  = (const float*)d_in[4];
    const float* encBih  = (const float*)d_in[5];
    const float* encBhh  = (const float*)d_in[6];
    const float* decWih  = (const float*)d_in[7];
    const float* decWhh  = (const float*)d_in[8];
    const float* decBih  = (const float*)d_in[9];
    const float* decBhh  = (const float*)d_in[10];
    const float* projW   = (const float*)d_in[11];
    const float* projB   = (const float*)d_in[12];
    float* out = (float*)d_out;

    char* ws = (char*)d_ws;
    size_t off = 0;
    auto alloc = [&](size_t bytes) {
        size_t o = off; off += (bytes + 255) & ~(size_t)255; return o;
    };
    _Float16* XencH   = (_Float16*)(ws + alloc((size_t)SEQ * HID * 2));
    _Float16* XdecH   = (_Float16*)(ws + alloc((size_t)SEQ * HID * 2));
    _Float16* WihEncH = (_Float16*)(ws + alloc((size_t)H3 * HID * 2));
    _Float16* WihDecH = (_Float16*)(ws + alloc((size_t)H3 * HID * 2));
    _Float16* ProjWH  = (_Float16*)(ws + alloc((size_t)VOCAB * HID * 2));
    float*    GXenc   = (float*)(ws + alloc((size_t)SEQ * H3 * 4));
    float*    GXdec   = (float*)(ws + alloc((size_t)SEQ * H3 * 4));
    _Float16* decHsH  = (_Float16*)(ws + alloc((size_t)SEQ * HID * 2));
    float*    hbuf    = (float*)(ws + alloc(2 * HID * 4));
    unsigned int* bar = (unsigned int*)(ws + alloc(256));

    (void)hipMemsetAsync(bar, 0, 256, stream);

    gather_embed<<<SEQ, 256, 0, stream>>>(inputs,  emb, XencH, 0);
    gather_embed<<<SEQ, 256, 0, stream>>>(targets, emb, XdecH, 1);
    {
        long n4 = (long)H3 * HID / 4;
        int g = (int)((n4 + 255) / 256);
        cvt_f32_f16<<<g, 256, 0, stream>>>(encWih, WihEncH, n4);
        cvt_f32_f16<<<g, 256, 0, stream>>>(decWih, WihDecH, n4);
        long p4 = (long)VOCAB * HID / 4;
        cvt_f32_f16<<<(int)((p4 + 255) / 256), 256, 0, stream>>>(projW, ProjWH, p4);
    }

    // GX = X @ Wih^T + bih   [512, 3072]
    gemm_wmma_f16<<<dim3(H3 / 256, SEQ / 64), 256, 0, stream>>>(
        XencH, WihEncH, encBih, GXenc, SEQ, H3, HID);
    gemm_wmma_f16<<<dim3(H3 / 256, SEQ / 64), 256, 0, stream>>>(
        XdecH, WihDecH, decBih, GXdec, SEQ, H3, HID);

    // Sequential recurrence: encoder then decoder, persistent grid
    gru_persistent<<<GRU_NWG, 256, 0, stream>>>(
        GXenc, GXdec, encWhh, decWhh, encBhh, decBhh, hbuf, decHsH, bar);

    // logits = dec_hs @ proj_W^T + proj_b   [512, 32000]
    gemm_wmma_f16<<<dim3(VOCAB / 256, SEQ / 64), 256, 0, stream>>>(
        decHsH, ProjWH, projB, out, SEQ, VOCAB, HID);
}
